// BPModel_83958020702517
// MI455X (gfx1250) — compile-verified, matrix operand order checked by reading, not run on hardware
//
#include <hip/hip_runtime.h>
#include <hip/hip_bf16.h>

typedef __bf16 bf16;
typedef __attribute__((ext_vector_type(16))) __bf16 v16bf;
typedef __attribute__((ext_vector_type(8)))  float  v8f;
typedef __attribute__((ext_vector_type(4)))  unsigned int u32x4;

#define B_TOT 4096
#define T_SEQ 256
#define HDIM  128
#define G4    512
#define LATD  128
#define NSTEP 9
#define DTC   (1.0f / 9.0f)

// ------------- WMMA fragment helper (ISA 7.12.2 16-bit layouts) -------------
// Per-lane data for both A (row-major MxK) and B stored transposed (N-major,
// NxK) is two contiguous 16-byte runs: rows r=lane&15, K offsets
// 8*half+{0..7} and 8*half+16+{0..7}.  ld must be a multiple of 8 elements.
__device__ inline v16bf frag16(const bf16* __restrict__ base, int ld) {
  int lane = threadIdx.x & 31;
  int half = lane >> 4;
  int r = lane & 15;
  const bf16* p = base + r * ld + 8 * half;
  union { u32x4 u[2]; v16bf f; } x;
  x.u[0] = *(const u32x4*)(p);
  x.u[1] = *(const u32x4*)(p + 16);
  return x.f;
}

__device__ inline v8f wmma_bf16(v16bf a, v16bf b, v8f c) {
  return __builtin_amdgcn_wmma_f32_16x16x32_bf16(false, a, false, b, (short)0, c,
                                                 false, false);
}

__device__ inline v8f v8f_zero() {
  v8f z;
#pragma unroll
  for (int i = 0; i < 8; ++i) z[i] = 0.f;
  return z;
}

// C/D layout: vgpr g, lane L -> row = g + 8*(L>>4), col = L&15
__device__ inline void store_D(float* __restrict__ out, int ldo, v8f acc) {
  int lane = threadIdx.x & 31;
  int half = lane >> 4;
  int c = lane & 15;
#pragma unroll
  for (int g = 0; g < 8; ++g) out[(g + 8 * half) * ldo + c] = acc[g];
}

__device__ inline void store_relu_bf16(bf16* __restrict__ out, int ldo, int row0,
                                       int col0, const float* __restrict__ bias,
                                       v8f acc) {
  int lane = threadIdx.x & 31;
  int half = lane >> 4;
  int c = lane & 15;
#pragma unroll
  for (int g = 0; g < 8; ++g) {
    int cc = col0 + c;
    float v = acc[g] + bias[cc];
    out[(row0 + g + 8 * half) * ldo + cc] = (bf16)(v > 0.f ? v : 0.f);
  }
}

__device__ inline void store_bias_f32(float* __restrict__ out, int ldo, int row0,
                                      int col0, const float* __restrict__ bias,
                                      v8f acc) {
  int lane = threadIdx.x & 31;
  int half = lane >> 4;
  int c = lane & 15;
#pragma unroll
  for (int g = 0; g < 8; ++g)
    out[(row0 + g + 8 * half) * ldo + col0 + c] = acc[g] + bias[col0 + c];
}

__device__ inline float sigmoid_(float x) { return 1.f / (1.f + __expf(-x)); }
__device__ inline float tanh_(float x) { return 2.f / (1.f + __expf(-2.f * x)) - 1.f; }

// ======================= Kernel A: LSTM encoder + fc =======================
// block = 256 threads (8 waves), 32 batch rows per block.
#define SMEM_A 211200

__global__ __launch_bounds__(256) void lstm_encoder_kernel(
    const float* __restrict__ x, const float* __restrict__ Wih,
    const float* __restrict__ Whh, const float* __restrict__ bih,
    const float* __restrict__ bhh, const float* __restrict__ fc1W,
    const float* __restrict__ fc1b, const float* __restrict__ fc2W,
    const float* __restrict__ fc2b, float* __restrict__ z0out) {
  extern __shared__ char smem[];
  bf16* sWhhT = (bf16*)(smem);                    // [512][128] bf16 (transposed)
  float* sG = (float*)(smem + 131072);            // 32*512 f32
  bf16* sH = (bf16*)(smem + 131072 + 65536);      // 32*128 bf16
  float* sWih = (float*)(smem + 204800);          // 2*512 f32
  float* sBias = (float*)(smem + 208896);         // 512 f32
  float* sXt = (float*)(smem + 210944);           // 64 f32 (x[:,t,:] tile)
  // fc-stage aliases (LSTM buffers dead by then)
  bf16* sWfc1T = (bf16*)(smem);                   // [256][128]
  bf16* sWfc2T = (bf16*)(smem + 65536);           // [128][256]
  bf16* sHid = (bf16*)(smem + 131072);            // 32*256 (aliases sG)
  float* sFb1 = sWih;                             // 256 f32
  float* sFb2 = sBias;                            // 128 f32

  const int tid = threadIdx.x;
  const int wave = tid >> 5;
  const int b0 = blockIdx.x * 32;

  {  // Whh [128][512] -> transposed bf16 LDS [n][k], coalesced float4 reads
    const float4* W4 = (const float4*)Whh;
    for (int i = tid; i < (HDIM * G4) / 4; i += 256) {
      float4 v = W4[i];
      int idx = i * 4, k = idx >> 9, n = idx & 511;
      sWhhT[(n + 0) * 128 + k] = (bf16)v.x;
      sWhhT[(n + 1) * 128 + k] = (bf16)v.y;
      sWhhT[(n + 2) * 128 + k] = (bf16)v.z;
      sWhhT[(n + 3) * 128 + k] = (bf16)v.w;
    }
  }
  for (int i = tid; i < 2 * G4; i += 256) sWih[i] = Wih[i];
  for (int i = tid; i < G4; i += 256) sBias[i] = bih[i] + bhh[i];
  for (int i = tid; i < 32 * HDIM; i += 256) sH[i] = (bf16)0.f;

  float creg[16];
#pragma unroll
  for (int q = 0; q < 16; ++q) creg[q] = 0.f;

  __syncthreads();

  // gate-phase invariants: j = tid & 127 is constant per thread
  const int jfix = tid & 127;
  const int mbase = tid >> 7;
  float wA[4], wB[4], bS[4];
#pragma unroll
  for (int e = 0; e < 4; ++e) {
    wA[e] = sWih[e * 128 + jfix];
    wB[e] = sWih[G4 + e * 128 + jfix];
    bS[e] = sBias[e * 128 + jfix];
  }

  const int mt = wave & 1;   // M tile (2 tiles of 16 rows)
  const int ng = wave >> 1;  // N tile group (4 groups of 8 tiles)

  for (int t = 0; t < T_SEQ; ++t) {
    // stage x[:,t,:] for this tile (read in gate phase after the barrier)
    if (tid < 64) sXt[tid] = x[((b0 + (tid >> 1)) * T_SEQ + t) * 2 + (tid & 1)];

    // ---- g = h @ Whh via WMMA ----
    v8f acc8[8];
#pragma unroll
    for (int u = 0; u < 8; ++u) acc8[u] = v8f_zero();
#pragma unroll
    for (int kk = 0; kk < 4; ++kk) {
      v16bf a = frag16(sH + mt * 16 * HDIM + kk * 32, HDIM);
#pragma unroll
      for (int u = 0; u < 8; ++u) {
        int nt = ng * 8 + u;
        v16bf b = frag16(sWhhT + nt * 16 * 128 + kk * 32, 128);
        acc8[u] = wmma_bf16(a, b, acc8[u]);
      }
    }
#pragma unroll
    for (int u = 0; u < 8; ++u)
      store_D(sG + mt * 16 * G4 + (ng * 8 + u) * 16, G4, acc8[u]);
    __syncthreads();

    // ---- gates ----
#pragma unroll
    for (int q = 0; q < 16; ++q) {
      int m = mbase + 2 * q;
      const float* g = sG + m * G4 + jfix;
      float x0 = sXt[2 * m], x1 = sXt[2 * m + 1];
      float gi = g[0]   + x0 * wA[0] + x1 * wB[0] + bS[0];
      float gf = g[128] + x0 * wA[1] + x1 * wB[1] + bS[1];
      float gg = g[256] + x0 * wA[2] + x1 * wB[2] + bS[2];
      float go = g[384] + x0 * wA[3] + x1 * wB[3] + bS[3];
      float cv = sigmoid_(gf) * creg[q] + sigmoid_(gi) * tanh_(gg);
      creg[q] = cv;
      sH[m * HDIM + jfix] = (bf16)(sigmoid_(go) * tanh_(cv));
    }
    __syncthreads();
  }

  // ---- fc1/fc2 on hN (weights transposed into LDS) ----
  {
    const float4* W4 = (const float4*)fc1W;  // [128][256]
    for (int i = tid; i < (128 * 256) / 4; i += 256) {
      float4 v = W4[i];
      int idx = i * 4, k = idx >> 8, n = idx & 255;
      sWfc1T[(n + 0) * 128 + k] = (bf16)v.x;
      sWfc1T[(n + 1) * 128 + k] = (bf16)v.y;
      sWfc1T[(n + 2) * 128 + k] = (bf16)v.z;
      sWfc1T[(n + 3) * 128 + k] = (bf16)v.w;
    }
    const float4* V4 = (const float4*)fc2W;  // [256][128]
    for (int i = tid; i < (256 * 128) / 4; i += 256) {
      float4 v = V4[i];
      int idx = i * 4, k = idx >> 7, n = idx & 127;
      sWfc2T[(n + 0) * 256 + k] = (bf16)v.x;
      sWfc2T[(n + 1) * 256 + k] = (bf16)v.y;
      sWfc2T[(n + 2) * 256 + k] = (bf16)v.z;
      sWfc2T[(n + 3) * 256 + k] = (bf16)v.w;
    }
  }
  if (tid < 256) sFb1[tid] = fc1b[tid];
  if (tid < 128) sFb2[tid] = fc2b[tid];
  __syncthreads();

  {  // fc1: M=32,N=256,K=128 ; 4 tiles per wave
    v8f acc4[4];
#pragma unroll
    for (int u = 0; u < 4; ++u) acc4[u] = v8f_zero();
#pragma unroll
    for (int kk = 0; kk < 4; ++kk) {
      v16bf a = frag16(sH + mt * 16 * HDIM + kk * 32, HDIM);
#pragma unroll
      for (int u = 0; u < 4; ++u) {
        int nt = ng * 4 + u;
        v16bf b = frag16(sWfc1T + nt * 16 * 128 + kk * 32, 128);
        acc4[u] = wmma_bf16(a, b, acc4[u]);
      }
    }
#pragma unroll
    for (int u = 0; u < 4; ++u)
      store_relu_bf16(sHid, 256, mt * 16, (ng * 4 + u) * 16, sFb1, acc4[u]);
  }
  __syncthreads();

  {  // fc2: M=32,N=128,K=256 ; 2 tiles per wave, write z0 to global
    v8f acc2[2];
    acc2[0] = v8f_zero();
    acc2[1] = v8f_zero();
#pragma unroll
    for (int kk = 0; kk < 8; ++kk) {
      v16bf a = frag16(sHid + mt * 16 * 256 + kk * 32, 256);
#pragma unroll
      for (int u = 0; u < 2; ++u) {
        int nt = ng * 2 + u;
        v16bf b = frag16(sWfc2T + nt * 16 * 256 + kk * 32, 256);
        acc2[u] = wmma_bf16(a, b, acc2[u]);
      }
    }
    int lane = tid & 31, half = lane >> 4, c = lane & 15;
#pragma unroll
    for (int u = 0; u < 2; ++u) {
      int nt = ng * 2 + u;
#pragma unroll
      for (int g = 0; g < 8; ++g) {
        int row = b0 + mt * 16 + g + 8 * half;
        int col = nt * 16 + c;
        z0out[row * LATD + col] = acc2[u][g] + sFb2[col];
      }
    }
  }
}

// ==================== Kernel B: RK4 latent ODE + decoder ====================
// block = 256 threads (8 waves), 16 batch rows per block.
// All weights stored TRANSPOSED (N-major, [N][K]) in LDS bf16.
#define OFF_PN1 0        // [64][128]
#define OFF_PN2 16384    // [128][64]
#define OFF_PN3 32768    // [16][128]  (N padded 3->16)
#define OFF_CN1 36864    // [64][128]
#define OFF_CN2 53248    // [128][64]
#define OFF_CN3 69632    // [128][128]
#define OFF_DE1 102400   // [128][160] (K padded 131->160)
#define OFF_DE2 143360   // [64][128]
#define OFF_DE3 159744   // [16][64]   (N padded 2->16)
#define OFF_BIAS 161792
#define OFF_Z    164864
#define OFF_ZIN  173056
#define OFF_K1   181248
#define OFF_K2   189440
#define OFF_K3   197632
#define OFF_K4   205824
#define OFF_COMP 214016
#define OFF_P    222208
#define OFF_PAR  223232
#define OFF_SA   223488
#define OFF_HP   228608
#define OFF_HC   230656
#define OFF_H2P  232704
#define OFF_H2C  236800
#define SMEM_B   240896

__global__ __launch_bounds__(256) void ode_decoder_kernel(
    const float* __restrict__ z0,
    const float* __restrict__ pn1W, const float* __restrict__ pn1b,
    const float* __restrict__ pn2W, const float* __restrict__ pn2b,
    const float* __restrict__ pn3W, const float* __restrict__ pn3b,
    const float* __restrict__ cn1W, const float* __restrict__ cn1b,
    const float* __restrict__ cn2W, const float* __restrict__ cn2b,
    const float* __restrict__ cn3W, const float* __restrict__ cn3b,
    const float* __restrict__ de1W, const float* __restrict__ de1b,
    const float* __restrict__ de2W, const float* __restrict__ de2b,
    const float* __restrict__ de3W, const float* __restrict__ de3b,
    float* __restrict__ out) {
  extern __shared__ char smem[];
  bf16* sPn1T = (bf16*)(smem + OFF_PN1);
  bf16* sPn2T = (bf16*)(smem + OFF_PN2);
  bf16* sPn3T = (bf16*)(smem + OFF_PN3);
  bf16* sCn1T = (bf16*)(smem + OFF_CN1);
  bf16* sCn2T = (bf16*)(smem + OFF_CN2);
  bf16* sCn3T = (bf16*)(smem + OFF_CN3);
  bf16* sDe1T = (bf16*)(smem + OFF_DE1);
  bf16* sDe2T = (bf16*)(smem + OFF_DE2);
  bf16* sDe3T = (bf16*)(smem + OFF_DE3);
  float* bAll = (float*)(smem + OFF_BIAS);
  float* bPn1 = bAll + 0;    float* bPn2 = bAll + 64;  float* bPn3 = bAll + 192;
  float* bCn1 = bAll + 208;  float* bCn2 = bAll + 272; float* bCn3 = bAll + 400;
  float* bDe1 = bAll + 528;  float* bDe2 = bAll + 656; float* bDe3 = bAll + 720;
  float* sZ = (float*)(smem + OFF_Z);
  float* sZin = (float*)(smem + OFF_ZIN);
  float* sK1 = (float*)(smem + OFF_K1);
  float* sK2 = (float*)(smem + OFF_K2);
  float* sK3 = (float*)(smem + OFF_K3);
  float* sK4 = (float*)(smem + OFF_K4);
  float* sComp = (float*)(smem + OFF_COMP);
  float* sP = (float*)(smem + OFF_P);
  float* sPar = (float*)(smem + OFF_PAR);
  bf16* sA = (bf16*)(smem + OFF_SA);    // [16][160] staging
  bf16* sHP = (bf16*)(smem + OFF_HP);   // [16][64]
  bf16* sHC = (bf16*)(smem + OFF_HC);   // [16][64]
  bf16* sH2P = (bf16*)(smem + OFF_H2P); // [16][128]
  bf16* sH2C = (bf16*)(smem + OFF_H2C); // [16][128]

  const int tid = threadIdx.x;
  const int wave = tid >> 5;
  const int b0 = blockIdx.x * 16;

  // ---- phase 1: zero padded weight regions + staging ----
  for (int i = tid; i < 16 * 128; i += 256) sPn3T[i] = (bf16)0.f;
  for (int i = tid; i < 128 * 160; i += 256) sDe1T[i] = (bf16)0.f;
  for (int i = tid; i < 16 * 64; i += 256) sDe3T[i] = (bf16)0.f;
  for (int i = tid; i < 16 * 160; i += 256) sA[i] = (bf16)0.f;
  __syncthreads();
  // ---- phase 2: fill transposed weights (source-major) + biases + z0 ----
  for (int i = tid; i < 128 * 64; i += 256) {
    int k = i >> 6, n = i & 63;
    sPn1T[n * 128 + k] = (bf16)pn1W[i];
    sCn1T[n * 128 + k] = (bf16)cn1W[i];
    sDe2T[n * 128 + k] = (bf16)de2W[i];
  }
  for (int i = tid; i < 64 * 128; i += 256) {
    int k = i >> 7, n = i & 127;
    sPn2T[n * 64 + k] = (bf16)pn2W[i];
    sCn2T[n * 64 + k] = (bf16)cn2W[i];
  }
  for (int i = tid; i < 128 * 3; i += 256) { int k = i / 3, n = i % 3; sPn3T[n * 128 + k] = (bf16)pn3W[i]; }
  for (int i = tid; i < 128 * 128; i += 256) { int k = i >> 7, n = i & 127; sCn3T[n * 128 + k] = (bf16)cn3W[i]; }
  for (int i = tid; i < 131 * 128; i += 256) { int k = i >> 7, n = i & 127; sDe1T[n * 160 + k] = (bf16)de1W[i]; }
  for (int i = tid; i < 64 * 2; i += 256) { int k = i >> 1, n = i & 1; sDe3T[n * 64 + k] = (bf16)de3W[i]; }
  if (tid < 64) { bPn1[tid] = pn1b[tid]; bCn1[tid] = cn1b[tid]; bDe2[tid] = de2b[tid]; }
  if (tid < 128) { bPn2[tid] = pn2b[tid]; bCn2[tid] = cn2b[tid]; bCn3[tid] = cn3b[tid]; bDe1[tid] = de1b[tid]; }
  if (tid < 16) { bPn3[tid] = tid < 3 ? pn3b[tid] : 0.f; bDe3[tid] = tid < 2 ? de3b[tid] : 0.f; }
  for (int i = tid; i < 16 * 128; i += 256) sZ[i] = z0[b0 * LATD + i];

  // odefunc MLP pair: param-net on waves 0-3, comp-net on waves 4-7.
  auto eval = [&](const float* zin) {
    __syncthreads();
    for (int i = tid; i < 16 * 128; i += 256) {
      int m = i >> 7, k = i & 127;
      sA[m * 160 + k] = (bf16)zin[i];
    }
    __syncthreads();
    {  // layer1: K=128 -> 64
      v16bf a[4];
#pragma unroll
      for (int kk = 0; kk < 4; ++kk) a[kk] = frag16(sA + kk * 32, 160);
      const bf16* W = (wave < 4) ? sPn1T : sCn1T;
      const float* bb = (wave < 4) ? bPn1 : bCn1;
      bf16* op = (wave < 4) ? sHP : sHC;
      int nt = wave & 3;
      v8f acc = v8f_zero();
#pragma unroll
      for (int kk = 0; kk < 4; ++kk)
        acc = wmma_bf16(a[kk], frag16(W + nt * 16 * 128 + kk * 32, 128), acc);
      store_relu_bf16(op, 64, 0, nt * 16, bb, acc);
    }
    __syncthreads();
    {  // layer2: K=64 -> 128, 16 tiles over 8 waves
#pragma unroll
      for (int u = 0; u < 2; ++u) {
        int id = wave * 2 + u;
        const bf16* in = (id < 8) ? sHP : sHC;
        const bf16* W = (id < 8) ? sPn2T : sCn2T;
        const float* bb = (id < 8) ? bPn2 : bCn2;
        bf16* op = (id < 8) ? sH2P : sH2C;
        int nt = id & 7;
        v8f acc = v8f_zero();
#pragma unroll
        for (int kk = 0; kk < 2; ++kk) {
          v16bf a = frag16(in + kk * 32, 64);
          acc = wmma_bf16(a, frag16(W + nt * 16 * 64 + kk * 32, 64), acc);
        }
        store_relu_bf16(op, 128, 0, nt * 16, bb, acc);
      }
    }
    __syncthreads();
    {  // layer3: comp = h2c @ cn3 (8 tiles); p = h2p @ pn3 (wave 0)
      v8f acc = v8f_zero();
#pragma unroll
      for (int kk = 0; kk < 4; ++kk) {
        v16bf a = frag16(sH2C + kk * 32, 128);
        acc = wmma_bf16(a, frag16(sCn3T + wave * 16 * 128 + kk * 32, 128), acc);
      }
      store_bias_f32(sComp, 128, 0, wave * 16, bCn3, acc);
      if (wave == 0) {
        v8f acc2 = v8f_zero();
#pragma unroll
        for (int kk = 0; kk < 4; ++kk) {
          v16bf a = frag16(sH2P + kk * 32, 128);
          acc2 = wmma_bf16(a, frag16(sPn3T + kk * 32, 128), acc2);
        }
        store_bias_f32(sP, 16, 0, 0, bPn3, acc2);
      }
    }
    __syncthreads();
  };

  auto dzcalc = [&](const float* zin, float* kout) {
#pragma unroll
    for (int q = 0; q < 8; ++q) {
      int i = tid + 256 * q;
      int m = i >> 7;
      float Rp = __expf(sP[m * 16 + 0]);
      float Rd = __expf(sP[m * 16 + 1]);
      float Cc = __expf(sP[m * 16 + 2]);
      float zv = zin[i];
      kout[i] = (-zv / Rd - Rp * zv + sComp[i]) / Cc;
    }
    __syncthreads();
  };

  // ---- params = exp(param_net(z0)) ----
  eval(sZ);
  if (tid < 16)
    for (int e = 0; e < 3; ++e) sPar[tid * 4 + e] = __expf(sP[tid * 16 + e]);
  __syncthreads();

  // ---- 9 fixed-grid RK4 (Kutta 3/8) steps ----
  for (int s = 0; s < NSTEP; ++s) {
    for (int q = 0; q < 8; ++q) { int i = tid + 256 * q; sZin[i] = sZ[i]; }
    eval(sZin); dzcalc(sZin, sK1);
    for (int q = 0; q < 8; ++q) { int i = tid + 256 * q; sZin[i] = sZ[i] + (DTC / 3.f) * sK1[i]; }
    eval(sZin); dzcalc(sZin, sK2);
    for (int q = 0; q < 8; ++q) { int i = tid + 256 * q; sZin[i] = sZ[i] + DTC * (sK2[i] - sK1[i] * (1.f / 3.f)); }
    eval(sZin); dzcalc(sZin, sK3);
    for (int q = 0; q < 8; ++q) { int i = tid + 256 * q; sZin[i] = sZ[i] + DTC * (sK1[i] - sK2[i] + sK3[i]); }
    eval(sZin); dzcalc(sZin, sK4);
    for (int q = 0; q < 8; ++q) {
      int i = tid + 256 * q;
      sZ[i] += DTC * 0.125f * (sK1[i] + 3.f * (sK2[i] + sK3[i]) + sK4[i]);
    }
    __syncthreads();
  }

  // ---- decoder on concat(zT, params) ----
  for (int i = tid; i < 16 * 128; i += 256) { int m = i >> 7, k = i & 127; sA[m * 160 + k] = (bf16)sZ[i]; }
  if (tid < 48) { int m = tid / 3, e = tid % 3; sA[m * 160 + 128 + e] = (bf16)sPar[m * 4 + e]; }
  __syncthreads();
  {  // dec1: K=160 (padded) -> 128, 1 tile per wave
    v8f acc = v8f_zero();
    for (int kk = 0; kk < 5; ++kk) {
      v16bf a = frag16(sA + kk * 32, 160);
      acc = wmma_bf16(a, frag16(sDe1T + wave * 16 * 160 + kk * 32, 160), acc);
    }
    store_relu_bf16(sH2P, 128, 0, wave * 16, bDe1, acc);
  }
  __syncthreads();
  if (wave < 4) {  // dec2: K=128 -> 64
    v8f acc = v8f_zero();
#pragma unroll
    for (int kk = 0; kk < 4; ++kk) {
      v16bf a = frag16(sH2P + kk * 32, 128);
      acc = wmma_bf16(a, frag16(sDe2T + wave * 16 * 128 + kk * 32, 128), acc);
    }
    store_relu_bf16(sHP, 64, 0, wave * 16, bDe2, acc);
  }
  __syncthreads();
  if (wave == 0) {  // dec3: K=64 -> 16 (2 valid cols)
    v8f acc = v8f_zero();
#pragma unroll
    for (int kk = 0; kk < 2; ++kk) {
      v16bf a = frag16(sHP + kk * 32, 64);
      acc = wmma_bf16(a, frag16(sDe3T + kk * 32, 64), acc);
    }
    store_bias_f32(sP, 16, 0, 0, bDe3, acc);
  }
  __syncthreads();
  if (tid < 32) {
    int m = tid >> 1, c = tid & 1;
    out[(b0 + m) * 2 + c] = sP[m * 16 + c];
  }
}

// ================================ launcher =================================
extern "C" void kernel_launch(void* const* d_in, const int* in_sizes, int n_in,
                              void* d_out, int out_size, void* d_ws, size_t ws_size,
                              hipStream_t stream) {
  const float* x = (const float*)d_in[0];
  const float* lstm_Wih = (const float*)d_in[1];
  const float* lstm_Whh = (const float*)d_in[2];
  const float* lstm_bih = (const float*)d_in[3];
  const float* lstm_bhh = (const float*)d_in[4];
  const float* fc1_W = (const float*)d_in[5];
  const float* fc1_b = (const float*)d_in[6];
  const float* fc2_W = (const float*)d_in[7];
  const float* fc2_b = (const float*)d_in[8];
  const float* pn1_W = (const float*)d_in[9];
  const float* pn1_b = (const float*)d_in[10];
  const float* pn2_W = (const float*)d_in[11];
  const float* pn2_b = (const float*)d_in[12];
  const float* pn3_W = (const float*)d_in[13];
  const float* pn3_b = (const float*)d_in[14];
  const float* cn1_W = (const float*)d_in[15];
  const float* cn1_b = (const float*)d_in[16];
  const float* cn2_W = (const float*)d_in[17];
  const float* cn2_b = (const float*)d_in[18];
  const float* cn3_W = (const float*)d_in[19];
  const float* cn3_b = (const float*)d_in[20];
  const float* dec1_W = (const float*)d_in[21];
  const float* dec1_b = (const float*)d_in[22];
  const float* dec2_W = (const float*)d_in[23];
  const float* dec2_b = (const float*)d_in[24];
  const float* dec3_W = (const float*)d_in[25];
  const float* dec3_b = (const float*)d_in[26];

  float* z0 = (float*)d_ws;  // [4096,128] f32 = 2 MB scratch
  float* out = (float*)d_out;

  (void)hipFuncSetAttribute((const void*)lstm_encoder_kernel,
                            hipFuncAttributeMaxDynamicSharedMemorySize, SMEM_A);
  (void)hipFuncSetAttribute((const void*)ode_decoder_kernel,
                            hipFuncAttributeMaxDynamicSharedMemorySize, SMEM_B);

  lstm_encoder_kernel<<<B_TOT / 32, 256, SMEM_A, stream>>>(
      x, lstm_Wih, lstm_Whh, lstm_bih, lstm_bhh, fc1_W, fc1_b, fc2_W, fc2_b, z0);

  ode_decoder_kernel<<<B_TOT / 16, 256, SMEM_B, stream>>>(
      z0, pn1_W, pn1_b, pn2_W, pn2_b, pn3_W, pn3_b, cn1_W, cn1_b, cn2_W, cn2_b,
      cn3_W, cn3_b, dec1_W, dec1_b, dec2_W, dec2_b, dec3_W, dec3_b, out);
}
